// FLASH_26268019982692
// MI455X (gfx1250) — matrix operand order repaired and checked by reading, not compiled, not run on hardware
//
#include <hip/hip_runtime.h>
#include <hip/hip_bf16.h>

// ---------------------------------------------------------------------------
// CDNA5 (gfx1250) FLASH/GAU block, bf16 WMMA everywhere.
// Global->LDS staging uses GLOBAL_LOAD_ASYNC_TO_LDS_B128 (ASYNCcnt).
// ---------------------------------------------------------------------------

typedef __bf16        bf16x16 __attribute__((ext_vector_type(16)));
typedef float         v8f     __attribute__((ext_vector_type(8)));
typedef unsigned int  uint4v  __attribute__((ext_vector_type(4)));

union FragBF {
    bf16x16        v;
    uint4v         q[2];
    unsigned short s[16];
};

__device__ __forceinline__ unsigned short f2bf(float f) {
    unsigned u = __float_as_uint(f);
    unsigned r = (u + 0x7FFFu + ((u >> 16) & 1u)) >> 16;   // RNE
    return (unsigned short)r;
}
__device__ __forceinline__ float bf2f(unsigned short h) {
    return __uint_as_float(((unsigned)h) << 16);
}

// LDS byte offset of a __shared__ object (addrspace(3) ptrtoint)
typedef const __attribute__((address_space(3))) void* lds_cptr_t;
__device__ __forceinline__ unsigned lds_addr(const void* p) {
    return (unsigned)(unsigned long long)(lds_cptr_t)p;
}

// Async DMA: 16 bytes per lane, global (SADDR + 32-bit VGPR offset) -> LDS.
__device__ __forceinline__ void async_load_b128(unsigned lds_off, unsigned voff,
                                                const void* sbase) {
    asm volatile("global_load_async_to_lds_b128 %0, %1, %2"
                 :: "v"(lds_off), "v"(voff), "s"(sbase)
                 : "memory");
}
__device__ __forceinline__ void wait_asynccnt0() {
    asm volatile("s_wait_asynccnt 0x0" ::: "memory");
}

// ---------------------------------------------------------------------------
// Elementwise f32 -> bf16 cast (weights)
// ---------------------------------------------------------------------------
__global__ void cast_bf16_kernel(const float* __restrict__ in,
                                 unsigned short* __restrict__ out, long n) {
    long i = (long)blockIdx.x * blockDim.x + threadIdx.x;
    long stride = (long)gridDim.x * blockDim.x;
    for (; i < n; i += stride) out[i] = f2bf(in[i]);
}

// ---------------------------------------------------------------------------
// LayerNorm over D=1024, one row per block (256 threads), bf16 output
// ---------------------------------------------------------------------------
__global__ __launch_bounds__(256) void ln_kernel(const float* __restrict__ x,
                                                 const float* __restrict__ g,
                                                 const float* __restrict__ b,
                                                 unsigned short* __restrict__ out) {
    const int tid = threadIdx.x;
    const long row = blockIdx.x;
    const float* xr = x + row * 1024;
    float vals[4], s = 0.f, s2 = 0.f;
    for (int i = 0; i < 4; ++i) {
        float v = xr[tid + 256 * i];
        vals[i] = v; s += v; s2 += v * v;
    }
    __shared__ float red0[256], red1[256];
    red0[tid] = s; red1[tid] = s2; __syncthreads();
    for (int st = 128; st > 0; st >>= 1) {
        if (tid < st) { red0[tid] += red0[tid + st]; red1[tid] += red1[tid + st]; }
        __syncthreads();
    }
    float mu  = red0[0] * (1.f / 1024.f);
    float var = red1[0] * (1.f / 1024.f) - mu * mu;
    float inv = rsqrtf(var + 1e-5f);
    for (int i = 0; i < 4; ++i) {
        int c = tid + 256 * i;
        out[row * 1024 + c] = f2bf((vals[i] - mu) * inv * g[c] + b[c]);
    }
}

// ---------------------------------------------------------------------------
// OffsetScale (4 heads) + RoPE over sequence dim. qk: [B*N,128] bf16.
// Heads out: quad_q, lin_q, quad_k, lin_k, each [B*N,128] bf16.
// ---------------------------------------------------------------------------
__global__ void rope_os_kernel(const unsigned short* __restrict__ qk,
                               const float* __restrict__ gamma,
                               const float* __restrict__ beta,
                               unsigned short* __restrict__ o0,
                               unsigned short* __restrict__ o1,
                               unsigned short* __restrict__ o2,
                               unsigned short* __restrict__ o3) {
    const long row = (long)blockIdx.x * blockDim.y + threadIdx.y;   // B*N rows
    const int  d   = threadIdx.x;                                   // 0..63
    const int  pos = (int)(row & 2047);                             // N = 2048
    float theta = (float)pos * __powf(10000.f, (float)d * (1.f / 64.f));
    float sn, cs; __sincosf(theta, &sn, &cs);
    const unsigned short* qr = qk + row * 128;
    float x1 = bf2f(qr[d]), x2 = bf2f(qr[d + 64]);
    unsigned short* outs[4] = { o0, o1, o2, o3 };
    #pragma unroll
    for (int h = 0; h < 4; ++h) {
        float y1 = x1 * gamma[h * 128 + d]      + beta[h * 128 + d];
        float y2 = x2 * gamma[h * 128 + d + 64] + beta[h * 128 + d + 64];
        unsigned short* o = outs[h] + row * 128;
        o[d]      = f2bf(y1 * cs - y2 * sn);
        o[d + 64] = f2bf(y2 * cs + y1 * sn);
    }
}

// ---------------------------------------------------------------------------
// Generic bf16 WMMA GEMM: C[z] = epilogue(A[z] x B[z])
// Block tile 128x64, K-step 64 (2 WMMA slices per stage), 8 wave32 waves,
// 2x2 16x16 WMMA tiles per wave. Non-transposed operands are staged with
// GLOBAL_LOAD_ASYNC_TO_LDS_B128; transposed fills gather via VGPRs.
// TA/TB: transpose applied while filling LDS. EPI:
//   0: silu(acc + bias)                -> bf16
//   1: relu(acc*scale)^2               -> bf16
//   2: acc*scale                       -> bf16
//   3: (acc + add_bf) * mul_bf         -> bf16
//   4: acc + bias + add_f32 (residual) -> f32
// ---------------------------------------------------------------------------
#define BM 128
#define BN 64
#define BKs 64

template <int EPI, int TA, int TB>
__global__ __launch_bounds__(256) void wmma_gemm_kernel(
    const unsigned short* __restrict__ A, long lda, long strideA,
    const unsigned short* __restrict__ Bm, long ldb, long strideB,
    unsigned short* __restrict__ Cb, float* __restrict__ Cf, long ldc, long strideC,
    const float* __restrict__ bias,
    const unsigned short* __restrict__ addb, long ld_add, long stride_add,
    const unsigned short* __restrict__ mulb, long ld_mul, long stride_mul,
    const float* __restrict__ addf, long ld_addf,
    float scale, int K) {

    __shared__ __align__(16) unsigned short As[BM][72];   // padded 16B-aligned rows
    __shared__ __align__(16) unsigned short Bs[BKs][72];

    const int tid  = threadIdx.x;
    const int lane = tid & 31;
    const int wave = tid >> 5;
    const int wm   = (wave & 3) * 32;   // wave M offset in block tile
    const int wn   = (wave >> 2) * 32;  // wave N offset in block tile
    const long z   = blockIdx.z;
    const long m0  = (long)blockIdx.y * BM;
    const long n0  = (long)blockIdx.x * BN;

    const unsigned short* Ab = A + z * strideA;
    const unsigned short* Bb = Bm + z * strideB;

    v8f zero = {};
    v8f acc[2][2];
    acc[0][0] = zero; acc[0][1] = zero; acc[1][0] = zero; acc[1][1] = zero;

    // Per-thread staging coordinates
    const int arow = tid >> 1, ac0 = (tid & 1) * 32;       // A: 128 rows x 64 cols
    const int brow = tid >> 2, bc0 = (tid & 3) * 16;       // B:  64 rows x 64 cols
    const unsigned a_lds = lds_addr(&As[arow][ac0]);
    const unsigned b_lds = lds_addr(&Bs[brow][bc0]);

    for (int kt = 0; kt < K; kt += BKs) {
        // ---- stage A tile (BM x BKs) ----
        if (TA == 0) {
            unsigned voff = (unsigned)(((m0 + arow) * lda + kt + ac0) * 2);
            #pragma unroll
            for (int u = 0; u < 4; ++u)
                async_load_b128(a_lds + u * 16, voff + u * 16, Ab);
        } else {
            #pragma unroll
            for (int i = 0; i < 32; ++i) {
                int idx = tid * 32 + i;
                int r = idx >> 6, c = idx & 63;
                As[r][c] = Ab[(long)(kt + c) * lda + m0 + r];
            }
        }
        // ---- stage B tile (BKs x BN) ----
        if (TB == 0) {
            unsigned voff = (unsigned)(((long)(kt + brow) * ldb + n0 + bc0) * 2);
            async_load_b128(b_lds,      voff,      Bb);
            async_load_b128(b_lds + 16, voff + 16, Bb);
        } else {
            #pragma unroll
            for (int i = 0; i < 16; ++i) {
                int idx = tid * 16 + i;
                int kk = idx & 63, nn = idx >> 6;
                Bs[kk][nn] = Bb[(long)(n0 + nn) * ldb + kt + kk];
            }
        }
        if (kt + BKs < K) {   // prefetch next tiles -> global_prefetch_b8
            __builtin_prefetch(Ab + (m0 + arow) * lda + kt + BKs, 0, 0);
            __builtin_prefetch(Bb + (long)(kt + BKs + brow) * ldb + n0, 0, 0);
        }
        if (TA == 0 || TB == 0) wait_asynccnt0();
        __syncthreads();

        // ---- 2 K-slices of WMMA per stage (ISA 7.12.2 fragment layouts) ----
        const int khi = (lane >> 4) << 3;     // lanes 16..31 take K+8 halves
        #pragma unroll
        for (int s = 0; s < 2; ++s) {
            FragBF afr[2], bfr[2];
            #pragma unroll
            for (int i = 0; i < 2; ++i) {
                int r = wm + 16 * i + (lane & 15);
                afr[i].q[0] = *(const uint4v*)&As[r][32 * s + khi];
                afr[i].q[1] = *(const uint4v*)&As[r][32 * s + 16 + khi];
            }
            #pragma unroll
            for (int j = 0; j < 2; ++j) {
                int c = wn + 16 * j;
                #pragma unroll
                for (int e = 0; e < 16; ++e)
                    bfr[j].s[e] = Bs[32 * s + (e & 8) * 2 + khi + (e & 7)][c + (lane & 15)];
            }
            #pragma unroll
            for (int i = 0; i < 2; ++i)
                #pragma unroll
                for (int j = 0; j < 2; ++j)
                    acc[i][j] = __builtin_amdgcn_wmma_f32_16x16x32_bf16(
                        false, afr[i].v, false, bfr[j].v,
                        (short)0, acc[i][j], false, false);
        }
        __syncthreads();
    }

    // ---- epilogue ----
    const int lm = lane & 15;
    const int ro = (lane >> 4) * 8;
    #pragma unroll
    for (int i = 0; i < 2; ++i)
        #pragma unroll
        for (int j = 0; j < 2; ++j)
            #pragma unroll
            for (int r = 0; r < 8; ++r) {
                long rg = m0 + wm + 16 * i + r + ro;
                long cg = n0 + wn + 16 * j + lm;
                float v = acc[i][j][r];
                if (EPI == 0) {
                    v += bias[cg];
                    v = v / (1.f + __expf(-v));
                    Cb[z * strideC + rg * ldc + cg] = f2bf(v);
                } else if (EPI == 1) {
                    v *= scale; v = v > 0.f ? v * v : 0.f;
                    Cb[z * strideC + rg * ldc + cg] = f2bf(v);
                } else if (EPI == 2) {
                    Cb[z * strideC + rg * ldc + cg] = f2bf(v * scale);
                } else if (EPI == 3) {
                    v = (v + bf2f(addb[z * stride_add + rg * ld_add + cg])) *
                        bf2f(mulb[z * stride_mul + rg * ld_mul + cg]);
                    Cb[z * strideC + rg * ldc + cg] = f2bf(v);
                } else {
                    v += bias[cg] + addf[rg * ld_addf + cg];
                    Cf[z * strideC + rg * ldc + cg] = v;
                }
            }
}

// ---------------------------------------------------------------------------
// Host-side orchestration
// ---------------------------------------------------------------------------
extern "C" void kernel_launch(void* const* d_in, const int* in_sizes, int n_in,
                              void* d_out, int out_size, void* d_ws, size_t ws_size,
                              hipStream_t stream) {
    (void)in_sizes; (void)n_in; (void)out_size; (void)ws_size;

    const float* x    = (const float*)d_in[0];
    const float* ln_g = (const float*)d_in[1];
    const float* ln_b = (const float*)d_in[2];
    const float* Wh   = (const float*)d_in[3];
    const float* bh   = (const float*)d_in[4];
    const float* Wqk  = (const float*)d_in[5];
    const float* bqk  = (const float*)d_in[6];
    const float* osg  = (const float*)d_in[7];
    const float* osb  = (const float*)d_in[8];
    const float* Wo   = (const float*)d_in[9];
    const float* bo   = (const float*)d_in[10];
    float* out = (float*)d_out;

    const long Bb = 4, Nn = 2048, Dd = 1024, Hh = 2048, QKd = 128;
    const long M  = Bb * Nn;            // 8192
    const long NG = 32;                 // B * (N/G) groups

    size_t off = 0;
    auto alloc = [&](size_t elems) -> unsigned short* {
        off = (off + 255) & ~(size_t)255;
        unsigned short* p = (unsigned short*)((char*)d_ws + off);
        off += elems * sizeof(unsigned short);
        return p;
    };
    unsigned short* normed  = alloc(M * Dd);          // bf16 LN output
    unsigned short* Whb     = alloc(Dd * 2 * Hh);
    unsigned short* Wqkb    = alloc(Dd * QKd);
    unsigned short* Wob     = alloc(Hh * Dd);
    unsigned short* hbuf    = alloc(M * 2 * Hh);      // [v | gate]
    unsigned short* qkbuf   = alloc(M * QKd);
    unsigned short* qq      = alloc(M * QKd);
    unsigned short* lq      = alloc(M * QKd);
    unsigned short* qkk     = alloc(M * QKd);
    unsigned short* lk      = alloc(M * QKd);
    unsigned short* attn    = alloc(NG * 256 * 256);
    unsigned short* quadout = alloc(M * Hh);
    unsigned short* linkv   = alloc(Bb * QKd * Hh);
    unsigned short* gated   = alloc(M * Hh);

    // 1) weights -> bf16
    cast_bf16_kernel<<<2048, 256, 0, stream>>>(Wh,  Whb,  Dd * 2 * Hh);
    cast_bf16_kernel<<<512,  256, 0, stream>>>(Wqk, Wqkb, Dd * QKd);
    cast_bf16_kernel<<<1024, 256, 0, stream>>>(Wo,  Wob,  Hh * Dd);

    // 2) LayerNorm
    ln_kernel<<<(int)M, 256, 0, stream>>>(x, ln_g, ln_b, normed);

    // 3) h = silu(normed @ Wh + bh)  [8192 x 4096]
    wmma_gemm_kernel<0, 0, 0><<<dim3(2 * Hh / 64, M / 128, 1), 256, 0, stream>>>(
        normed, Dd, 0, Whb, 2 * Hh, 0, hbuf, nullptr, 2 * Hh, 0,
        bh, nullptr, 0, 0, nullptr, 0, 0, nullptr, 0, 1.f, (int)Dd);

    // 4) qk = silu(normed @ Wqk + bqk)  [8192 x 128]
    wmma_gemm_kernel<0, 0, 0><<<dim3(QKd / 64, M / 128, 1), 256, 0, stream>>>(
        normed, Dd, 0, Wqkb, QKd, 0, qkbuf, nullptr, QKd, 0,
        bqk, nullptr, 0, 0, nullptr, 0, 0, nullptr, 0, 1.f, (int)Dd);

    // 5) OffsetScale + RoPE -> 4 heads
    rope_os_kernel<<<(int)(M / 4), dim3(64, 4), 0, stream>>>(qkbuf, osg, osb,
                                                             qq, lq, qkk, lk);

    // 6) attn = relu(qq @ qk^T / G)^2   per group [32 x 256 x 256]
    wmma_gemm_kernel<1, 0, 1><<<dim3(256 / 64, 256 / 128, (int)NG), 256, 0, stream>>>(
        qq, QKd, 256 * QKd, qkk, QKd, 256 * QKd, attn, nullptr, 256, 256 * 256,
        nullptr, nullptr, 0, 0, nullptr, 0, 0, nullptr, 0, 1.f / 256.f, (int)QKd);

    // 7) quad_out = attn @ v   per group [32 x 256 x 2048]
    wmma_gemm_kernel<2, 0, 0><<<dim3(Hh / 64, 256 / 128, (int)NG), 256, 0, stream>>>(
        attn, 256, 256 * 256, hbuf, 2 * Hh, 256 * 2 * Hh, quadout, nullptr, Hh, 256 * Hh,
        nullptr, nullptr, 0, 0, nullptr, 0, 0, nullptr, 0, 1.f, 256);

    // 8) lin_kv = (lin_k^T @ v) / N   per batch [4 x 128 x 2048]
    wmma_gemm_kernel<2, 1, 0><<<dim3(Hh / 64, 1, (int)Bb), 256, 0, stream>>>(
        lk, QKd, Nn * QKd, hbuf, 2 * Hh, Nn * 2 * Hh, linkv, nullptr, Hh, QKd * Hh,
        nullptr, nullptr, 0, 0, nullptr, 0, 0, nullptr, 0, 1.f / (float)Nn, (int)Nn);

    // 9) gated = gate * (lin_q @ lin_kv + quad_out)   per batch [4 x 2048 x 2048]
    wmma_gemm_kernel<3, 0, 0><<<dim3(Hh / 64, Nn / 128, (int)Bb), 256, 0, stream>>>(
        lq, QKd, Nn * QKd, linkv, Hh, QKd * Hh, gated, nullptr, Hh, Nn * Hh,
        nullptr, quadout, Hh, Nn * Hh, hbuf + Hh, 2 * Hh, Nn * 2 * Hh,
        nullptr, 0, 1.f, (int)QKd);

    // 10) out = gated @ Wo + bo + x   [8192 x 1024] f32
    wmma_gemm_kernel<4, 0, 0><<<dim3(Dd / 64, M / 128, 1), 256, 0, stream>>>(
        gated, Hh, 0, Wob, Dd, 0, nullptr, out, Dd, 0,
        bo, nullptr, 0, 0, nullptr, 0, 0, x, Dd, 1.f, (int)Hh);
}
